// ConvSDF_17016660427429
// MI455X (gfx1250) — matrix operand order repaired and checked by reading, not compiled, not run on hardware
//
#include <hip/hip_runtime.h>
#include <hip/hip_bf16.h>
#include <math.h>

typedef __attribute__((ext_vector_type(16))) _Float16 v16h;
typedef __attribute__((ext_vector_type(8)))  _Float16 v8h;
typedef __attribute__((ext_vector_type(8)))  float    v8f;

#define B_      8
#define N_      16384
#define M_      4
#define OUTC_   32
#define NCELLS_ 27
#define KPAD_   32
#define DIL_    0.05f
#define MAXD_   1.0f
#define ROWS_   16
#define WAVES_  4
#define TILES_PER_B_ (N_ / ROWS_)              // 1024
#define BLOCKS_PER_B_ (TILES_PER_B_ / WAVES_)  // 256

__global__ __launch_bounds__(WAVES_ * 32) void convsdf_fused(
    const float* __restrict__ locs,  const int* __restrict__ idxs,
    const float* __restrict__ poses, const float* __restrict__ scales,
    const float* __restrict__ weight, const float* __restrict__ bias,
    const float* __restrict__ sdfs,  const int* __restrict__ sdf_offsets,
    const float* __restrict__ sdf_shapes, float* __restrict__ out)
{
    __shared__ __align__(16) _Float16 Atile[WAVES_][ROWS_][KPAD_];

    const int lane = threadIdx.x & 31;
    const int w    = threadIdx.x >> 5;

    // b is block-uniform -> per-object descriptors scalarize to SGPRs.
    const int b  = blockIdx.x / BLOCKS_PER_B_;
    const int n0 = (blockIdx.x % BLOCKS_PER_B_) * (WAVES_ * ROWS_) + w * ROWS_;
    const int row0 = b * N_ + n0;   // flattened (b,n) row base for this wave

    // ---- Preload the M=4 object descriptors (uniform -> scalar loads) ----
    float otx[M_], oty[M_], otz[M_];          // translation
    float oux[M_], ouy[M_], ouz[M_], oqw[M_]; // -q.xyz, q.w
    float os[M_], oinv[M_];                   // scale, 1/(scale*cell)
    float odx1[M_], ody1[M_], odz1[M_];       // dims-1 (float bounds)
    int   odx2[M_], ody2[M_], odz2[M_];       // dims-2 (i0 clamp)
    int   oisy[M_], oisz[M_];                 // strides
    long  obase[M_];                          // sdf base offset
    bool  ok[M_];
    #pragma unroll
    for (int m = 0; m < M_; ++m) {
        const int sidx = idxs[b * M_ + m];
        ok[m] = (sidx >= 0);
        const int sm = ok[m] ? sidx : 0;
        const float* ps = poses + (size_t)(b * M_ + m) * 7;
        otx[m] = ps[0]; oty[m] = ps[1]; otz[m] = ps[2];
        oux[m] = -ps[3]; ouy[m] = -ps[4]; ouz[m] = -ps[5]; oqw[m] = ps[6];
        const float s = scales[b * M_ + m];
        os[m] = s;
        const float* shp = sdf_shapes + (size_t)sm * 4;
        const float dx = shp[0], dy = shp[1], dz = shp[2], cl = shp[3];
        oinv[m] = 1.f / (s * cl);
        odx1[m] = dx - 1.f; ody1[m] = dy - 1.f; odz1[m] = dz - 1.f;
        odx2[m] = (int)dx - 2; ody2[m] = (int)dy - 2; odz2[m] = (int)dz - 2;
        oisy[m] = (int)dy; oisz[m] = (int)dz;
        obase[m] = (long)sdf_offsets[sm];
    }

    // ---- Phase 1: each lane owns one K column (cell) across the 16 rows ----
    const int cell = lane;              // lanes 27..31 supply zero padding
    float offx = 0.f, offy = 0.f, offz = 0.f;
    if (cell < NCELLS_) {
        offx = (float)((cell / 9)     - 1) * DIL_;
        offy = (float)((cell / 3) % 3 - 1) * DIL_;
        offz = (float)((cell % 3)     - 1) * DIL_;
    }

    #pragma unroll 4
    for (int r = 0; r < ROWS_; ++r) {
        float val = 0.f;
        if (cell < NCELLS_) {
            const float* lp = locs + ((size_t)b * N_ + (size_t)(n0 + r)) * 4;
            const float px = lp[0] + offx;
            const float py = lp[1] + offy;
            const float pz = lp[2] + offz;
            float best = __builtin_inff();
            #pragma unroll
            for (int m = 0; m < M_; ++m) {
                if (!ok[m]) continue;
                const float vx = px - otx[m], vy = py - oty[m], vz = pz - otz[m];
                // inverse quat rotation: t = 2*cross(u,v); rv = v + w*t + cross(u,t)
                const float ux = oux[m], uy = ouy[m], uz = ouz[m], qw = oqw[m];
                const float tx = 2.f * (uy * vz - uz * vy);
                const float ty = 2.f * (uz * vx - ux * vz);
                const float tz = 2.f * (ux * vy - uy * vx);
                const float rx = vx + qw * tx + (uy * tz - uz * ty);
                const float ry = vy + qw * ty + (uz * tx - ux * tz);
                const float rz = vz + qw * tz + (ux * ty - uy * tx);
                // grid coords: g = rv/(s*cell) - 0.5
                const float gx = rx * oinv[m] - 0.5f;
                const float gy = ry * oinv[m] - 0.5f;
                const float gz = rz * oinv[m] - 0.5f;
                const bool valid = (gx >= 0.f) && (gx <= odx1[m]) &&
                                   (gy >= 0.f) && (gy <= ody1[m]) &&
                                   (gz >= 0.f) && (gz <= odz1[m]);
                if (!valid) continue;
                // valid => gc == g; only the dims-2 upper clamp matters
                const int i0x = min((int)floorf(gx), odx2[m]);
                const int i0y = min((int)floorf(gy), ody2[m]);
                const int i0z = min((int)floorf(gz), odz2[m]);
                const float fx = gx - (float)i0x;
                const float fy = gy - (float)i0y;
                const float fz = gz - (float)i0z;

                const int isz = oisz[m];
                const float* g0 = sdfs + obase[m] +
                                  ((long)i0x * oisy[m] + i0y) * (long)isz + i0z;
                const float* g1 = g0 + (long)oisy[m] * isz;
                const float c000 = g0[0],   c001 = g0[1];
                const float c010 = g0[isz], c011 = g0[isz + 1];
                const float c100 = g1[0],   c101 = g1[1];
                const float c110 = g1[isz], c111 = g1[isz + 1];

                const float c00 = c000 + fz * (c001 - c000);
                const float c01 = c010 + fz * (c011 - c010);
                const float c10 = c100 + fz * (c101 - c100);
                const float c11 = c110 + fz * (c111 - c110);
                const float e0  = c00 + fy * (c01 - c00);
                const float e1  = c10 + fy * (c11 - c10);
                const float v   = e0 + fx * (e1 - e0);
                best = fminf(best, v * os[m]);
            }
            val = fminf(best, MAXD_);   // all-invalid -> inf -> MAX_DIST
        }
        Atile[w][r][lane] = (_Float16)val;
    }
    __syncthreads();

    // ---- Phase 2: wave WMMA, out[16 rows][32 outc] = A(16x32) * B(32x16) x2 ----
    // A fragment (contiguous K order, same as B): lane L -> row L%16;
    // halves 0..15 = K kb..kb+15, kb = (L<16)?0:16
    const int arow = lane & 15;
    const int kb   = (lane < 16) ? 0 : 16;
    const _Float16* ap = &Atile[w][arow][0];
    const v8h lo = *(const v8h*)(ap + kb);
    const v8h hi = *(const v8h*)(ap + kb + 8);
    v16h a;
    #pragma unroll
    for (int j = 0; j < 8; ++j) { a[j] = lo[j]; a[8 + j] = hi[j]; }

    // B fragments: lane L -> column o = tile*16 + (L%16); K = (L<16 ? 0..15 : 16..31)
    const int col   = lane & 15;
    const int krow0 = (lane < 16) ? 0 : 16;
    v16h b0, b1;
    #pragma unroll
    for (int j = 0; j < 16; ++j) {
        const int k = krow0 + j;
        const bool inK = (k < NCELLS_);
        b0[j] = inK ? (_Float16)weight[(size_t)col        * NCELLS_ + k] : (_Float16)0.f;
        b1[j] = inK ? (_Float16)weight[(size_t)(16 + col) * NCELLS_ + k] : (_Float16)0.f;
    }

    v8f acc0 = {}, acc1 = {};
    acc0 = __builtin_amdgcn_wmma_f32_16x16x32_f16(false, a, false, b0,
                                                  (short)0, acc0, false, false);
    acc1 = __builtin_amdgcn_wmma_f32_16x16x32_f16(false, a, false, b1,
                                                  (short)0, acc1, false, false);

    // C/D layout: VGPR r, lanes 0-15 -> row r, lanes 16-31 -> row r+8; col = lane%16
    const int mh = (lane < 16) ? 0 : 8;
    const float bia0 = bias[col];
    const float bia1 = bias[16 + col];
    #pragma unroll
    for (int r = 0; r < 8; ++r) {
        float* op = out + (size_t)(row0 + r + mh) * OUTC_;
        op[col]      = acc0[r] + bia0;
        op[16 + col] = acc1[r] + bia1;
    }
}

extern "C" void kernel_launch(void* const* d_in, const int* in_sizes, int n_in,
                              void* d_out, int out_size, void* d_ws, size_t ws_size,
                              hipStream_t stream) {
    const float* locs        = (const float*)d_in[0];
    const int*   idxs        = (const int*)  d_in[1];
    const float* poses       = (const float*)d_in[2];
    const float* scales      = (const float*)d_in[3];
    const float* weight      = (const float*)d_in[4];
    const float* bias        = (const float*)d_in[5];
    const float* sdfs        = (const float*)d_in[6];
    const int*   sdf_offsets = (const int*)  d_in[7];
    const float* sdf_shapes  = (const float*)d_in[8];
    float* out = (float*)d_out;

    const int blocks = B_ * BLOCKS_PER_B_;   // 2048 blocks x 4 waves x 16-row tiles
    hipLaunchKernelGGL(convsdf_fused, dim3(blocks), dim3(WAVES_ * 32), 0, stream,
                       locs, idxs, poses, scales, weight, bias,
                       sdfs, sdf_offsets, sdf_shapes, out);
}
